// PlasticPolicy_40638980555358
// MI455X (gfx1250) — matrix operand order repaired and checked by reading, not compiled, not run on hardware
//
#include <hip/hip_runtime.h>
#include <hip/hip_bf16.h>

// ---------------------------------------------------------------------------
// Types for CDNA5 WMMA (wave32): V_WMMA_F32_16X16X32_BF16
//   A: 16x32 bf16 (8 VGPRs), B: 32x16 bf16 (8 VGPRs), C/D: 16x16 f32 (8 VGPRs)
// ---------------------------------------------------------------------------
typedef __attribute__((ext_vector_type(16))) __bf16 v16bf;
typedef __attribute__((ext_vector_type(8)))  float  v8f;

union FragU { v16bf v; uint4 q[2]; };
union Pk8   { uint4 q; __bf16 h[8]; };

// Load one 8-VGPR bf16 fragment as two 16-byte chunks at element offsets
// e0 and e1 from a *uniform* base pointer. Per-lane offsets are 32-bit ints,
// so the backend can emit saddr-form global_load_b128 (SGPR base + VGPR
// offset) with no per-iteration VALU address math.
//
// A-matrix 16x32 fragment (row-major A, lane = row r, half h):
//   VGPR0..3 hold K=2v+8h     -> elements off+8h   .. off+8h+7   (chunk e0)
//   VGPR4..7 hold K=16+2v'+8h -> elements off+16+8h.. off+23+8h  (chunk e0+16)
// B-matrix 32x16 fragment from Bt [N,K] (lane = col n, half h):
//   same K order as A         -> elements off+8h..off+8h+7, off+16+8h..off+23+8h  (e0, e0+16)
__device__ __forceinline__ v16bf ld_frag(const __bf16* __restrict__ base,
                                         int e0, int e1) {
  FragU f;
  f.q[0] = *(const uint4*)(base + e0);
  f.q[1] = *(const uint4*)(base + e1);
  return f.v;
}

// ---------------------------------------------------------------------------
// Generic bf16 WMMA GEMM:  out[z] = act( A[z] (MxK, lda) * Bt[z]^T (N x K) + bias[z] )
// Block: 256 threads (8 waves) arranged RW x CW; wave tile (16*TM) x (16*TN).
// Block tile: (RW*TM*16) x (CW*TN*16).  Requires M,N multiples of block tile,
// K % 32 == 0.  act: 0 none, 1 leaky(0.01), 2 relu, 3 clip[-20, 2]
// ---------------------------------------------------------------------------
template <int TM, int TN, int RW, int CW>
__global__ void __launch_bounds__(256)
gemm_bf16_wmma(const __bf16* __restrict__ A, long sAz, int lda,
               const __bf16* __restrict__ Bt, long sBz,
               const float* __restrict__ bias, long sbz,
               __bf16* __restrict__ obf, float* __restrict__ of32, long sOz,
               int N, int K, int act)
{
  const int z = blockIdx.z;
  A  += (long)z * sAz;
  Bt += (long)z * sBz;
  if (bias) bias += (long)z * sbz;

  const int lane = threadIdx.x & 31;
  const int wave = threadIdx.x >> 5;
  const int wm = wave / CW;
  const int wn = wave % CW;
  const int m0 = blockIdx.y * (RW * TM * 16) + wm * (TM * 16);
  const int n0 = blockIdx.x * (CW * TN * 16) + wn * (TN * 16);
  const int r  = lane & 15;
  const int h  = lane >> 4;

  // Per-lane 32-bit element offsets (loop-invariant -> stay in VGPRs; the
  // uniform base pointers advance scalar-side each K-step).
  int aoff[TM], boff[TN];
#pragma unroll
  for (int i = 0; i < TM; ++i) aoff[i] = (m0 + 16 * i + r) * lda + 8 * h;
#pragma unroll
  for (int j = 0; j < TN; ++j) boff[j] = (n0 + 16 * j + r) * K + 8 * h;

  const __bf16* __restrict__ Ak  = A;
  const __bf16* __restrict__ Btk = Bt;

  v8f acc[TM][TN] = {};
  for (int k = 0; k < K; k += 32) {
    v16bf fa[TM], fb[TN];
#pragma unroll
    for (int i = 0; i < TM; ++i) fa[i] = ld_frag(Ak, aoff[i], aoff[i] + 16);
#pragma unroll
    for (int j = 0; j < TN; ++j) fb[j] = ld_frag(Btk, boff[j], boff[j] + 16);
#pragma unroll
    for (int i = 0; i < TM; ++i)
#pragma unroll
      for (int j = 0; j < TN; ++j)
        acc[i][j] = __builtin_amdgcn_wmma_f32_16x16x32_bf16(
            false, fa[i], false, fb[j], (short)0, acc[i][j], false, false);
    Ak  += 32;   // uniform scalar advance
    Btk += 32;
  }

  if (obf)  obf  += (long)z * sOz;
  if (of32) of32 += (long)z * sOz;

  // C/D layout: lanes 0-15: (M = reg, N = lane); lanes 16-31: (M = 8+reg, N = lane-16)
#pragma unroll
  for (int i = 0; i < TM; ++i) {
#pragma unroll
    for (int j = 0; j < TN; ++j) {
      const int gm0 = m0 + i * 16 + h * 8;
      const int gn  = n0 + j * 16 + r;
      const float bb = bias ? bias[gn] : 0.0f;
#pragma unroll
      for (int t = 0; t < 8; ++t) {
        float v = acc[i][j][t] + bb;
        if      (act == 1) v = (v >= 0.0f) ? v : 0.01f * v;
        else if (act == 2) v = fmaxf(v, 0.0f);
        else if (act == 3) v = fminf(fmaxf(v, -20.0f), 2.0f);
        const long o = (long)(gm0 + t) * N + gn;
        if (obf)  obf[o]  = (__bf16)v;
        if (of32) of32[o] = v;
      }
    }
  }
}

// ---------------------------------------------------------------------------
// BatchNorm stats, stage 1: per (e, slice s) partial sum/sumsq over B/S rows.
// 256 threads: n = tid%64 (coalesced across N), 4 b-groups.
// ---------------------------------------------------------------------------
__global__ void __launch_bounds__(256)
bn_partial(const float* __restrict__ states, float* __restrict__ psum,
           float* __restrict__ psq, int B, int N, int S)
{
  const int e = blockIdx.y, s = blockIdx.x;
  const int n = threadIdx.x & 63;
  const int g = threadIdx.x >> 6;        // 0..3
  const int chunk = B / S;
  const float* base = states + ((long)e * B + (long)s * chunk) * N + n;
  float sum = 0.0f, sq = 0.0f;
  for (int b = g; b < chunk; b += 4) {
    float v = base[(long)b * N];
    sum += v; sq += v * v;
  }
  __shared__ float ls[256], lq[256];
  ls[threadIdx.x] = sum; lq[threadIdx.x] = sq;
  __syncthreads();
  if (g == 0) {
    sum = ls[n] + ls[n + 64] + ls[n + 128] + ls[n + 192];
    sq  = lq[n] + lq[n + 64] + lq[n + 128] + lq[n + 192];
    psum[((long)e * S + s) * N + n] = sum;
    psq [((long)e * S + s) * N + n] = sq;
  }
}

__global__ void bn_final(const float* __restrict__ psum, const float* __restrict__ psq,
                         float* __restrict__ mu, float* __restrict__ rs,
                         int B, int N, int S)
{
  const int t = threadIdx.x;            // one thread per (e,n), blockDim = E*N
  const int e = t >> 6, n = t & 63;
  float sum = 0.0f, sq = 0.0f;
  for (int s = 0; s < S; ++s) {
    sum += psum[((long)e * S + s) * N + n];
    sq  += psq [((long)e * S + s) * N + n];
  }
  const float m = sum / (float)B;
  const float var = sq / (float)B - m * m;   // biased variance (torch BN)
  mu[t] = m;
  rs[t] = rsqrtf(var + 1e-5f);
}

// s_norm (bf16) = (states - mu) * rsigma
__global__ void k_norm(const float* __restrict__ states, const float* __restrict__ mu,
                       const float* __restrict__ rs, __bf16* __restrict__ snorm,
                       long total, int BN, int N)
{
  long idx = (long)blockIdx.x * blockDim.x + threadIdx.x;
  if (idx >= total) return;
  const int e = (int)(idx / BN);
  const int n = (int)(idx % N);
  const int c = e * N + n;
  snorm[idx] = (__bf16)((states[idx] - mu[c]) * rs[c]);
}

// dst[z][c][r] = (bf16) src[z][r][c]   (weight transpose+convert; tiny matrices)
__global__ void tpack(const float* __restrict__ src, __bf16* __restrict__ dst,
                      int R, int C)
{
  const int z = blockIdx.z;
  const long n = (long)R * C;
  long idx = (long)blockIdx.x * blockDim.x + threadIdx.x;
  if (idx >= n) return;
  const int r = (int)(idx / C), c = (int)(idx % C);
  dst[(long)z * n + (long)c * R + r] = (__bf16)src[(long)z * n + idx];
}

__global__ void cvt_f32_bf16(const float* __restrict__ src, __bf16* __restrict__ dst, long n)
{
  long idx = (long)blockIdx.x * blockDim.x + threadIdx.x;
  if (idx < n) dst[idx] = (__bf16)src[idx];
}

// ---------------------------------------------------------------------------
// Attention: one batch row per lane, one head per blockIdx.y.
// logits[e] = dot(sel[a,b,:], keys[e,b,a*64..]) / 8 ; softmax over E=8;
// actor_in[b, a*64+i] = sum_e w[e] * vals[e,b,a*64+i]
// ---------------------------------------------------------------------------
__global__ void __launch_bounds__(256)
attn_kernel(const __bf16* __restrict__ sel, const __bf16* __restrict__ keys,
            const __bf16* __restrict__ vals, float* __restrict__ actor_f32,
            __bf16* __restrict__ actor_bf, int B)
{
  const int a = blockIdx.y;
  const int b = blockIdx.x * blockDim.x + threadIdx.x;
  if (b >= B) return;

  float s[64];
  {
    const uint4* p = (const uint4*)(sel + (long)b * 512 + a * 64);
#pragma unroll
    for (int c = 0; c < 8; ++c) {
      Pk8 u; u.q = p[c];
#pragma unroll
      for (int j = 0; j < 8; ++j) s[c * 8 + j] = (float)u.h[j];
    }
  }

  float logit[8];
#pragma unroll
  for (int e = 0; e < 8; ++e) {
    const uint4* p = (const uint4*)(keys + ((long)e * B + b) * 512 + a * 64);
    float acc = 0.0f;
#pragma unroll
    for (int c = 0; c < 8; ++c) {
      Pk8 u; u.q = p[c];
#pragma unroll
      for (int j = 0; j < 8; ++j) acc += s[c * 8 + j] * (float)u.h[j];
    }
    logit[e] = acc * 0.125f;              // 1/sqrt(d), d = 64
  }

  float mx = logit[0];
#pragma unroll
  for (int e = 1; e < 8; ++e) mx = fmaxf(mx, logit[e]);
  float w[8], sum = 0.0f;
#pragma unroll
  for (int e = 0; e < 8; ++e) { w[e] = __expf(logit[e] - mx); sum += w[e]; }
  const float inv = 1.0f / sum;

  float o[64];
#pragma unroll
  for (int i = 0; i < 64; ++i) o[i] = 0.0f;
#pragma unroll
  for (int e = 0; e < 8; ++e) {
    const float we = w[e] * inv;
    const uint4* p = (const uint4*)(vals + ((long)e * B + b) * 512 + a * 64);
#pragma unroll
    for (int c = 0; c < 8; ++c) {
      Pk8 u; u.q = p[c];
#pragma unroll
      for (int j = 0; j < 8; ++j) o[c * 8 + j] += we * (float)u.h[j];
    }
  }

  float*  op = actor_f32 + (long)b * 512 + a * 64;
  __bf16* ob = actor_bf  + (long)b * 512 + a * 64;
#pragma unroll
  for (int i = 0; i < 64; ++i) { op[i] = o[i]; ob[i] = (__bf16)o[i]; }
}

// ---------------------------------------------------------------------------
// Host-side orchestration
// ---------------------------------------------------------------------------
extern "C" void kernel_launch(void* const* d_in, const int* in_sizes, int n_in,
                              void* d_out, int out_size, void* d_ws, size_t ws_size,
                              hipStream_t stream) {
  const int E = 8, B = 16384, N = 64, H = 512;  // A=8 heads, d=64, A*d = H
  const int S = 64;                              // BN partial slices

  const float* states      = (const float*)d_in[0];
  const float* pre_actions = (const float*)d_in[1];
  const float* W_enc       = (const float*)d_in[2];
  const float* b_enc       = (const float*)d_in[3];
  const float* W_pre       = (const float*)d_in[4];
  const float* b_pre       = (const float*)d_in[5];
  const float* Wk          = (const float*)d_in[6];
  const float* Wsel        = (const float*)d_in[7];
  const float* Wv          = (const float*)d_in[8];
  const float* bv          = (const float*)d_in[9];   // [A,d] contiguous == packed [512] bias
  const float* W_actor     = (const float*)d_in[10];
  const float* b_actor     = (const float*)d_in[11];
  const float* W_mean      = (const float*)d_in[12];
  const float* b_mean      = (const float*)d_in[13];
  const float* W_logstd    = (const float*)d_in[14];
  const float* b_logstd    = (const float*)d_in[15];

  float* out_mean   = (float*)d_out;
  float* out_logstd = out_mean + (long)B * N;
  float* out_actor  = out_logstd + (long)B * N;

  // workspace carve-up (256B aligned)
  char* ws = (char*)d_ws;
  size_t off = 0;
  auto alloc = [&](size_t bytes) -> char* {
    off = (off + 255) & ~(size_t)255;
    char* p = ws + off;
    off += bytes;
    return p;
  };
  float*  psum    = (float*)alloc((size_t)E * S * N * 4);
  float*  psq     = (float*)alloc((size_t)E * S * N * 4);
  float*  mu      = (float*)alloc((size_t)E * N * 4);
  float*  rs      = (float*)alloc((size_t)E * N * 4);
  __bf16* snorm   = (__bf16*)alloc((size_t)E * B * N * 2);
  __bf16* preBf   = (__bf16*)alloc((size_t)B * H * 2);
  __bf16* WtEnc   = (__bf16*)alloc((size_t)E * H * N * 2);   // per e: [H, N]
  __bf16* WtPre   = (__bf16*)alloc((size_t)H * H * 2);       // [H, H]
  __bf16* WtSel   = (__bf16*)alloc((size_t)H * H * 2);       // [A*d, H]
  __bf16* WtK     = (__bf16*)alloc((size_t)H * H * 2);       // [A*d, H]
  __bf16* WtV     = (__bf16*)alloc((size_t)H * H * 2);       // [A*d, H]
  __bf16* WtAct   = (__bf16*)alloc((size_t)H * H * 2);       // [H, H]
  __bf16* WtMean  = (__bf16*)alloc((size_t)N * H * 2);       // [N, H]
  __bf16* WtLog   = (__bf16*)alloc((size_t)N * H * 2);
  __bf16* preEnc  = (__bf16*)alloc((size_t)B * H * 2);
  __bf16* selBuf  = (__bf16*)alloc((size_t)B * H * 2);
  __bf16* encs    = (__bf16*)alloc((size_t)E * B * H * 2);
  __bf16* keysAll = (__bf16*)alloc((size_t)E * B * H * 2);
  __bf16* valsAll = (__bf16*)alloc((size_t)E * B * H * 2);
  __bf16* actorBf = (__bf16*)alloc((size_t)B * H * 2);
  __bf16* xBf     = (__bf16*)alloc((size_t)B * H * 2);
  (void)ws_size; (void)in_sizes; (void)n_in; (void)out_size;

  // 1) BatchNorm statistics
  bn_partial<<<dim3(S, E), 256, 0, stream>>>(states, psum, psq, B, N, S);
  bn_final<<<1, E * N, 0, stream>>>(psum, psq, mu, rs, B, N, S);

  // 2) normalize states -> bf16
  {
    long total = (long)E * B * N;
    k_norm<<<dim3((unsigned)((total + 255) / 256)), 256, 0, stream>>>(
        states, mu, rs, snorm, total, B * N, N);
  }

  // 3) weight repack to bf16 transposed layouts
  tpack<<<dim3((N * H + 255) / 256, 1, E), 256, 0, stream>>>(W_enc,   WtEnc, N, H);
  tpack<<<dim3((H * H + 255) / 256, 1, 1), 256, 0, stream>>>(W_pre,   WtPre, H, H);
  tpack<<<dim3((H * 64 + 255) / 256, 1, E), 256, 0, stream>>>(Wsel,   WtSel, H, 64);
  tpack<<<dim3((H * 64 + 255) / 256, 1, E), 256, 0, stream>>>(Wk,     WtK,   H, 64);
  tpack<<<dim3((H * 64 + 255) / 256, 1, E), 256, 0, stream>>>(Wv,     WtV,   H, 64);
  tpack<<<dim3((H * H + 255) / 256, 1, 1), 256, 0, stream>>>(W_actor, WtAct, H, H);
  tpack<<<dim3((H * N + 255) / 256, 1, 1), 256, 0, stream>>>(W_mean,  WtMean, H, N);
  tpack<<<dim3((H * N + 255) / 256, 1, 1), 256, 0, stream>>>(W_logstd, WtLog, H, N);
  cvt_f32_bf16<<<dim3((unsigned)(((long)B * H + 255) / 256)), 256, 0, stream>>>(
      pre_actions, preBf, (long)B * H);

  const dim3 blk(256);
  // Wide GEMMs (N = 512): wave tile 64x64, block tile 128x256, 16 wmma / k-step.
  // 4) encs[e] = leaky(s_norm[e] @ W_enc[e] + b_enc[e])   M=B, N=H, K=64
  gemm_bf16_wmma<4, 4, 2, 4><<<dim3(H / 256, B / 128, E), blk, 0, stream>>>(
      snorm, (long)B * 64, 64, WtEnc, (long)H * 64, b_enc, H,
      encs, nullptr, (long)B * H, H, 64, 1);
  // 5) pre = leaky(pre_actions @ W_pre + b_pre)   M=B, N=H, K=H
  gemm_bf16_wmma<4, 4, 2, 4><<<dim3(H / 256, B / 128, 1), blk, 0, stream>>>(
      preBf, 0, H, WtPre, 0, b_pre, 0, preEnc, nullptr, 0, H, H, 1);
  // 6) sel = pre @ Wsel (all heads)   M=B, N=A*d=H, K=H
  gemm_bf16_wmma<4, 4, 2, 4><<<dim3(H / 256, B / 128, 1), blk, 0, stream>>>(
      preEnc, 0, H, WtSel, 0, nullptr, 0, selBuf, nullptr, 0, H, H, 0);
  // 7) keys[e] = encs[e] @ Wk (all heads)   M=B, N=H, K=H, batched over E
  gemm_bf16_wmma<4, 4, 2, 4><<<dim3(H / 256, B / 128, E), blk, 0, stream>>>(
      encs, (long)B * H, H, WtK, 0, nullptr, 0,
      keysAll, nullptr, (long)B * H, H, H, 0);
  // 8) vals[e] = leaky(encs[e] @ Wv + bv)
  gemm_bf16_wmma<4, 4, 2, 4><<<dim3(H / 256, B / 128, E), blk, 0, stream>>>(
      encs, (long)B * H, H, WtV, 0, bv, 0,
      valsAll, nullptr, (long)B * H, H, H, 1);
  // 9) attention -> actor_in (f32 out + bf16 feed)
  attn_kernel<<<dim3(B / 256, 8), 256, 0, stream>>>(
      selBuf, keysAll, valsAll, out_actor, actorBf, B);
  // 10) x = relu(actor_in @ W_actor + b_actor)
  gemm_bf16_wmma<4, 4, 2, 4><<<dim3(H / 256, B / 128, 1), blk, 0, stream>>>(
      actorBf, 0, H, WtAct, 0, b_actor, 0, xBf, nullptr, 0, H, H, 2);
  // Narrow GEMMs (N = 64): wave tile 32x32, block tile 128x64.
  // 11) mean = x @ W_mean + b_mean   (f32 out)
  gemm_bf16_wmma<2, 2, 4, 2><<<dim3(N / 64, B / 128, 1), blk, 0, stream>>>(
      xBf, 0, H, WtMean, 0, b_mean, 0, nullptr, out_mean, 0, N, H, 0);
  // 12) log_std = clip(x @ W_logstd + b_logstd, -20, 2)
  gemm_bf16_wmma<2, 2, 4, 2><<<dim3(N / 64, B / 128, 1), blk, 0, stream>>>(
      xBf, 0, H, WtLog, 0, b_logstd, 0, nullptr, out_logstd, 0, N, H, 3);
}
